// OtaMatcher_47974784697138
// MI455X (gfx1250) — matrix operand order repaired and checked by reading, not compiled, not run on hardware
//
#include <hip/hip_runtime.h>
#include <cmath>

#define B_  16
#define Q_  1000
#define G_  300
#define C_  80
#define CP_ 96   // C padded to 3 k-steps of 32 for bf16 WMMA

typedef __attribute__((ext_vector_type(16))) __bf16 v16bf;
typedef __attribute__((ext_vector_type(8)))  __bf16 v8bf;
typedef __attribute__((ext_vector_type(8)))  float  v8f;

__device__ __forceinline__ float softplusf(float x) {
    // stable softplus: max(x,0) + log1p(exp(-|x|))
    return fmaxf(x, 0.0f) + log1pf(__expf(-fabsf(x)));
}

__device__ __forceinline__ float rcpf(float x) {
    return __builtin_amdgcn_rcpf(x);   // v_rcp_f32, ~1 ulp
}

// ---------------------------------------------------------------------------
// Kernel 1: per (b,q) focal-loss prep.
//   delta2[b][q][c] = COST_CLASS * (loss_pos - loss_neg), stored as bf16
//   hi/lo pair (transposed to [q][Cpad] so WMMA B-matrix loads are contiguous
//   per lane), negsum2[b][q] = COST_CLASS * sum_c loss_neg, and the
//   background row of the output: out[b][G][q] = COST_CLASS * neg_sum.
// ---------------------------------------------------------------------------
__global__ void ota_prep_kernel(const float* __restrict__ logits,
                                __bf16* __restrict__ dhi,
                                __bf16* __restrict__ dlo,
                                float* __restrict__ negsum2,
                                float* __restrict__ out)
{
    int tid = blockIdx.x * blockDim.x + threadIdx.x;
    if (tid >= B_ * Q_) return;
    int b = tid / Q_;
    int q = tid - b * Q_;

    const float* lp  = logits + ((size_t)b * Q_ + q) * C_;
    __bf16*      hp  = dhi + ((size_t)b * Q_ + q) * CP_;
    __bf16*      lop = dlo + ((size_t)b * Q_ + q) * CP_;

    float ns = 0.0f;
    #pragma unroll 4
    for (int c = 0; c < C_; ++c) {
        float x  = lp[c];
        float p  = rcpf(1.0f + __expf(-x));   // sigmoid
        float ce_pos = softplusf(-x);
        float ce_neg = softplusf(x);
        float lpos = 0.25f * (1.0f - p) * (1.0f - p) * ce_pos;
        float lneg = 0.75f * p * p * ce_neg;
        ns += lneg;
        float d  = 2.0f * (lpos - lneg);      // fold COST_CLASS (exact x2)
        __bf16 h = (__bf16)d;
        hp[c]  = h;
        lop[c] = (__bf16)(d - (float)h);      // residual for hi/lo split
    }
    #pragma unroll
    for (int c = C_; c < CP_; ++c) { hp[c] = (__bf16)0.0f; lop[c] = (__bf16)0.0f; }

    float ns2 = 2.0f * ns;                    // fold COST_CLASS (exact x2)
    negsum2[tid] = ns2;
    // background row (g == G): cost = COST_CLASS * neg_sum
    out[(size_t)b * (G_ + 1) * Q_ + (size_t)G_ * Q_ + q] = ns2;
}

// ---------------------------------------------------------------------------
// Kernel 2: pair costs. Block = 128 threads = 4 waves, all sharing one
// 16-row g-tile; each wave owns a 16(g) x 16(q) tile. The class-cost gather
// is onehot(gt_classes) @ delta via V_WMMA_F32_16X16X32_BF16 (hi+lo).
// The onehot A matrix (16 x 96 bf16) is built once per block in LDS and each
// wave pulls its per-lane fragment with two aligned ds_load_b128 per k-step.
// The same wave then fuses L1 + GIoU (rcp-based) and streams the fg rows.
// ---------------------------------------------------------------------------
__global__ __launch_bounds__(128)
void ota_pair_kernel(const float* __restrict__ pred_boxes,
                     const float* __restrict__ gt_boxes,
                     const float* __restrict__ img,       // (B,4)
                     const float* __restrict__ img_tgt,   // (B,G,4)
                     const int*   __restrict__ gt_classes,
                     const __bf16* __restrict__ dhi,
                     const __bf16* __restrict__ dlo,
                     const float* __restrict__ negsum2,
                     float* __restrict__ out)
{
    const int b    = blockIdx.z;
    const int gb   = blockIdx.y * 16;
    const int tid  = threadIdx.x;
    const int wave = tid >> 5;
    const int lane = tid & 31;
    const int qb   = (blockIdx.x * 4 + wave) * 16;
    const int n    = lane & 15;          // q column within tile / A row m
    const int hi   = lane >> 4;          // lane half selects K sub-block

    // Shared onehot A (16 rows x 96 cols bf16) + normalized GT tile (+area).
    __shared__ __align__(16) __bf16 s_a[16 * CP_];
    __shared__ float s_tgt[16][5];

    // Zero-fill A with b128 stores spread over the block (192 x 16B chunks).
    {
        int4 z = {0, 0, 0, 0};
        int4* sa4 = (int4*)s_a;
        for (int i = tid; i < (16 * CP_) / 8; i += 128) sa4[i] = z;
    }
    __syncthreads();

    if (tid < 16) {
        int g = min(gb + tid, G_ - 1);
        // onehot: one bf16 1.0 per row at that row's class
        int cls = gt_classes[(size_t)b * G_ + g];
        s_a[tid * CP_ + cls] = (__bf16)1.0f;
        // normalized gt box + precomputed area
        const float* tb = gt_boxes + ((size_t)b * G_ + g) * 4;
        const float* is = img_tgt  + ((size_t)b * G_ + g) * 4;
        float t0 = tb[0] * rcpf(is[0]);
        float t1 = tb[1] * rcpf(is[1]);
        float t2 = tb[2] * rcpf(is[2]);
        float t3 = tb[3] * rcpf(is[3]);
        s_tgt[tid][0] = t0;
        s_tgt[tid][1] = t1;
        s_tgt[tid][2] = t2;
        s_tgt[tid][3] = t3;
        s_tgt[tid][4] = (t2 - t0) * (t3 - t1);
    }
    __syncthreads();

    // B-matrix: lane n = q column; per k-step a contiguous 16-element run.
    const int q = min(qb + n, Q_ - 1);
    const __bf16* bh = dhi + ((size_t)b * Q_ + q) * CP_ + (hi ? 16 : 0);
    const __bf16* bl = dlo + ((size_t)b * Q_ + q) * CP_ + (hi ? 16 : 0);

    // A fragment source: this lane's row, one contiguous 16-element run per k-step.
    const __bf16* arow = s_a + (lane & 15) * CP_ + hi * 16;

    v8f acc = {};
    #pragma unroll
    for (int kt = 0; kt < 3; ++kt) {
        const int kbase = kt * 32;
        // A 16x32 bf16 layout: lane elements = rows m, element e ->
        // k = kbase + e + (hi ? 16 : 0): one contiguous 32B run.
        v16bf a = *(const v16bf*)(arow + kbase);
        v16bf vbh = *(const v16bf*)(bh + kbase);
        v16bf vbl = *(const v16bf*)(bl + kbase);
        acc = __builtin_amdgcn_wmma_f32_16x16x32_bf16(false, a, false, vbh,
                                                      (short)0, acc, false, false);
        acc = __builtin_amdgcn_wmma_f32_16x16x32_bf16(false, a, false, vbl,
                                                      (short)0, acc, false, false);
    }
    // acc[r] = COST_CLASS * gathered delta for (g = gb + r + 8*hi, q = qb + n)

    // Per-lane predicted box (fixed q), normalized via rcp multiply.
    float o0, o1, o2, o3, areab, nsq;
    {
        const float* pb = pred_boxes + ((size_t)b * Q_ + q) * 4;
        const float* is = img + (size_t)b * 4;
        o0 = pb[0] * rcpf(is[0]);
        o1 = pb[1] * rcpf(is[1]);
        o2 = pb[2] * rcpf(is[2]);
        o3 = pb[3] * rcpf(is[3]);
        areab = (o2 - o0) * (o3 - o1);
        nsq   = negsum2[(size_t)b * Q_ + q];
    }

    const bool qok = (qb + n) < Q_;
    #pragma unroll
    for (int r = 0; r < 8; ++r) {
        const int rr = r + (hi ? 8 : 0);
        const int g  = gb + rr;
        const float t0 = s_tgt[rr][0], t1 = s_tgt[rr][1];
        const float t2 = s_tgt[rr][2], t3 = s_tgt[rr][3];
        const float areaa = s_tgt[rr][4];

        const float ltx = fmaxf(t0, o0), lty = fmaxf(t1, o1);
        const float rbx = fminf(t2, o2), rby = fminf(t3, o3);
        const float wx  = fmaxf(rbx - ltx, 0.0f);
        const float wy  = fmaxf(rby - lty, 0.0f);
        const float inter = wx * wy;
        const float uni   = areaa + areab - inter;
        const float iou   = inter * rcpf(uni);
        const float ex = fmaxf(fmaxf(t2, o2) - fminf(t0, o0), 0.0f);
        const float ey = fmaxf(fmaxf(t3, o3) - fminf(t1, o1), 0.0f);
        const float enc  = ex * ey;
        const float giou = iou - (enc - uni) * rcpf(enc);
        const float l1 = fabsf(t0 - o0) + fabsf(t1 - o1) +
                         fabsf(t2 - o2) + fabsf(t3 - o3);

        // fg = COST_CLASS*(negsum + gathered) + COST_BBOX*l1 + COST_GIOU*(-giou)
        const float fg = (nsq + acc[r]) + 5.0f * l1 - 2.0f * giou;
        if (qok && g < G_)
            out[(size_t)b * (G_ + 1) * Q_ + (size_t)g * Q_ + (qb + n)] = fg;
    }
}

extern "C" void kernel_launch(void* const* d_in, const int* in_sizes, int n_in,
                              void* d_out, int out_size, void* d_ws, size_t ws_size,
                              hipStream_t stream) {
    const float* pred_logits = (const float*)d_in[0];
    const float* pred_boxes  = (const float*)d_in[1];
    const float* gt_boxes    = (const float*)d_in[2];
    const float* img         = (const float*)d_in[3];
    const float* img_tgt     = (const float*)d_in[4];
    const int*   gt_classes  = (const int*)d_in[5];
    float* out = (float*)d_out;

    __bf16* dhi     = (__bf16*)d_ws;
    __bf16* dlo     = dhi + (size_t)B_ * Q_ * CP_;
    float*  negsum2 = (float*)(dlo + (size_t)B_ * Q_ * CP_);
    // workspace use: 2*16*1000*96*2 B + 16*1000*4 B ~= 6.2 MB

    const int nprep = B_ * Q_;
    ota_prep_kernel<<<(nprep + 255) / 256, 256, 0, stream>>>(
        pred_logits, dhi, dlo, negsum2, out);

    dim3 grid((Q_ + 63) / 64, (G_ + 15) / 16, B_);
    ota_pair_kernel<<<grid, 128, 0, stream>>>(
        pred_boxes, gt_boxes, img, img_tgt, gt_classes,
        dhi, dlo, negsum2, out);
}
